// ResidualAttentionBlock_55533927138037
// MI455X (gfx1250) — compile-verified, matrix operand order checked by reading, not run on hardware
//
#include <hip/hip_runtime.h>
#include <hip/hip_bf16.h>

// ---------------------------------------------------------------------------
// CDNA5 (gfx1250, wave32) ResidualAttentionBlock + ST-adapter.
// All dense matmuls -> v_wmma_f32_16x16x32_bf16 (fp32 cast to bf16, fp32 acc).
// Depthwise 3D convs / layernorm / softmax are VALU.
// ---------------------------------------------------------------------------

typedef __attribute__((ext_vector_type(16))) __bf16 v16bf;
typedef __attribute__((ext_vector_type(8)))  __bf16 v8bf;
typedef __attribute__((ext_vector_type(4)))  __bf16 v4bf;
typedef __attribute__((ext_vector_type(8)))  float  v8f;

#define D_MODEL 1024
#define N_HEAD  16
#define HDIM    64
#define LSEQ    577
#define BT      32
#define CB      4      // batch for conv
#define CT      8      // frames
#define CHW     24     // H = W = 24
#define CA      128    // adapter width
#define LM1     576    // L-1
#define M_AD    (BT * LM1)          // 18432 adapter GEMM rows
#define M_FULL  (BT * LSEQ)         // 18464 full-seq GEMM rows
#define NCONV   (CB * CA * CT * CHW * CHW)  // 2,359,296

// Combine two aligned 8-element bf16 LDS chunks into one WMMA fragment.
__device__ __forceinline__ v16bf frag_from_lds(const __bf16* p, int khalf)
{
    const v8bf lo = *reinterpret_cast<const v8bf*>(p + khalf);        // 16B aligned
    const v8bf hi = *reinterpret_cast<const v8bf*>(p + 16 + khalf);   // 16B aligned
    return __builtin_shufflevector(lo, hi, 0, 1, 2, 3, 4, 5, 6, 7,
                                   8, 9, 10, 11, 12, 13, 14, 15);
}

__device__ __forceinline__ void cvt4_scaled(v16bf& a, int o, float4 v, float s)
{
    a[o + 0] = (__bf16)(v.x * s);
    a[o + 1] = (__bf16)(v.y * s);
    a[o + 2] = (__bf16)(v.z * s);
    a[o + 3] = (__bf16)(v.w * s);
}

__device__ __forceinline__ v4bf pack4(float4 v)
{
    v4bf r;
    r[0] = (__bf16)v.x; r[1] = (__bf16)v.y; r[2] = (__bf16)v.z; r[3] = (__bf16)v.w;
    return r;
}

// ---------------------------------------------------------------------------
// Generic WMMA GEMM:  C[M,N] = act(A[M,K] * W[N,K]^T + bias[N]) (+ residual)
// 256 threads = 8 waves; block tile 128(M) x 64(N); K-step 32.
// Wave grid 4x2, each wave owns a 32x32 output tile -> 4 wmma per K-step.
// AMAP: 0 = A row m at A + m*K ; 1 = adapter map (skip CLS token, ld=1024)
// SMAP: 0 = C idx m*N+n       ; 1 = adapter scatter into [*,577,1024]
// ACT : 0 = none ; 1 = QuickGELU
// RES : 0/1 add residual (same mapped index as the store)
// ---------------------------------------------------------------------------
template <int AMAP, int SMAP, int ACT, int RES>
__global__ void __launch_bounds__(256)
gemm_wmma_kernel(const float* __restrict__ A, const float* __restrict__ W,
                 const float* __restrict__ bias, const float* __restrict__ res,
                 float* __restrict__ C, int M, int N, int K)
{
    __shared__ __bf16 As[128][40];   // 80B rows: every 16B chunk stays aligned
    __shared__ __bf16 Bs[64][40];

    const int bm   = blockIdx.y;
    const int bn   = blockIdx.x;
    const int tid  = threadIdx.x;
    const int lane = tid & 31;
    const int wv   = tid >> 5;
    const int wm   = wv & 3;              // 4 waves along M
    const int wn   = wv >> 2;             // 2 waves along N
    const int khalf = (lane >> 4) * 8;    // ISA 7.12.2 bf16 A/B lane layout

    v8f acc[2][2] = {};

    const int acol  = (tid & 7) * 4;      // 8 thread-groups cover K=32 as float4
    const int arow0 = tid >> 3;           // 0..31

    for (int k0 = 0; k0 < K; k0 += 32) {
        // ---- stage A tile (128x32 f32 -> bf16 LDS) ----
#pragma unroll
        for (int rr = 0; rr < 4; ++rr) {
            const int r  = arow0 + rr * 32;
            const int gm = bm * 128 + r;
            float4 av = make_float4(0.f, 0.f, 0.f, 0.f);
            if (gm < M) {
                size_t rowoff;
                if (AMAP == 1)
                    rowoff = (size_t)((gm / LM1) * LSEQ + 1 + (gm % LM1)) * D_MODEL;
                else
                    rowoff = (size_t)gm * K;
                av = *reinterpret_cast<const float4*>(A + rowoff + k0 + acol);
            }
            *reinterpret_cast<v4bf*>(&As[r][acol]) = pack4(av);
        }
        // ---- stage W tile (64x32 f32 -> bf16 LDS) ----
#pragma unroll
        for (int rr = 0; rr < 2; ++rr) {
            const int r  = arow0 + rr * 32;
            const int gn = bn * 64 + r;
            const float4 wvv =
                *reinterpret_cast<const float4*>(W + (size_t)gn * K + k0 + acol);
            *reinterpret_cast<v4bf*>(&Bs[r][acol]) = pack4(wvv);
        }
        if (k0 + 32 < K)   // gfx1250 global_prefetch hint for next W tile
            __builtin_prefetch((const void*)(W + (size_t)(bn * 64) * K + k0 + 32), 0, 1);
        __syncthreads();

        // ---- build fragments (ds_load_b128 pairs) & 4 WMMAs ----
        v16bf afrag[2], bfrag[2];
#pragma unroll
        for (int ms = 0; ms < 2; ++ms)
            afrag[ms] = frag_from_lds(&As[wm * 32 + ms * 16 + (lane & 15)][0], khalf);
#pragma unroll
        for (int ns = 0; ns < 2; ++ns)
            bfrag[ns] = frag_from_lds(&Bs[wn * 32 + ns * 16 + (lane & 15)][0], khalf);
#pragma unroll
        for (int ms = 0; ms < 2; ++ms)
#pragma unroll
            for (int ns = 0; ns < 2; ++ns)
                acc[ms][ns] = __builtin_amdgcn_wmma_f32_16x16x32_bf16(
                    false, afrag[ms], false, bfrag[ns],
                    (short)0, acc[ms][ns], false, false);
        __syncthreads();
    }

    // ---- epilogue: bias / activation / residual / scatter ----
#pragma unroll
    for (int ms = 0; ms < 2; ++ms)
#pragma unroll
        for (int ns = 0; ns < 2; ++ns) {
            const int gn = bn * 64 + wn * 32 + ns * 16 + (lane & 15);
            const float bv = bias[gn];
#pragma unroll
            for (int i = 0; i < 8; ++i) {
                const int ml = wm * 32 + ms * 16 + i + ((lane & 16) ? 8 : 0);
                const int gm = bm * 128 + ml;
                if (gm >= M) continue;
                float v = acc[ms][ns][i] + bv;
                if (ACT == 1)  // QuickGELU
                    v = v / (1.f + __expf(-1.702f * v));
                size_t idx;
                if (SMAP == 1)
                    idx = (size_t)((gm / LM1) * LSEQ + 1 + (gm % LM1)) * D_MODEL + gn;
                else
                    idx = (size_t)gm * N + gn;
                if (RES) v += res[idx];
                C[idx] = v;
            }
        }
}

// ---------------------------------------------------------------------------
// Attention: one WG (128 thr = 4 waves) per (b, head, 16-row q tile).
// S = (Q/8)·K^T via WMMA into LDS, wave-parallel softmax, O = P·V via WMMA.
// All global loads are unconditional with clamped row indices (garbage rows
// are masked out by the softmax column bound / zero pad / guarded stores).
// ---------------------------------------------------------------------------
__global__ void __launch_bounds__(128)
attn_kernel(const float* __restrict__ qkv, float* __restrict__ o)
{
    __shared__ float sS[16][616];   // 16 q-rows x (577 padded to 616)

    const int qt = blockIdx.x, h = blockIdx.y, b = blockIdx.z;
    const int l0 = qt * 16;
    const int wv = threadIdx.x >> 5, lane = threadIdx.x & 31;
    const int khalf = (lane >> 4) * 8;
    const float scale = 0.125f;     // 1/sqrt(64)

    // ---- Q fragments (16 x 64, two K-halves of 32), clamped rows ----
    v16bf qa[2];
    {
        const int mrow = min(l0 + (lane & 15), LSEQ - 1);
        const float* qb = qkv + (size_t)(b * LSEQ + mrow) * 3 * D_MODEL + h * HDIM;
#pragma unroll
        for (int ks = 0; ks < 2; ++ks) {
            const float4 q0 = *reinterpret_cast<const float4*>(qb + ks * 32 + khalf);
            const float4 q1 = *reinterpret_cast<const float4*>(qb + ks * 32 + khalf + 4);
            const float4 q2 = *reinterpret_cast<const float4*>(qb + ks * 32 + 16 + khalf);
            const float4 q3 = *reinterpret_cast<const float4*>(qb + ks * 32 + 20 + khalf);
            v16bf a;
            cvt4_scaled(a, 0,  q0, scale);
            cvt4_scaled(a, 4,  q1, scale);
            cvt4_scaled(a, 8,  q2, scale);
            cvt4_scaled(a, 12, q3, scale);
            qa[ks] = a;
        }
    }

    // ---- Phase 1: S = Q K^T, 16-col tiles round-robined over 4 waves ----
    for (int nt = wv; nt * 16 < LSEQ; nt += 4) {
        const int n0   = nt * 16;
        const int nrow = min(n0 + (lane & 15), LSEQ - 1);
        const float* kb =
            qkv + (size_t)(b * LSEQ + nrow) * 3 * D_MODEL + (N_HEAD + h) * HDIM;
        v8f acc = {};
#pragma unroll
        for (int ks = 0; ks < 2; ++ks) {
            const float4 k0v = *reinterpret_cast<const float4*>(kb + ks * 32 + khalf);
            const float4 k1v = *reinterpret_cast<const float4*>(kb + ks * 32 + khalf + 4);
            const float4 k2v = *reinterpret_cast<const float4*>(kb + ks * 32 + 16 + khalf);
            const float4 k3v = *reinterpret_cast<const float4*>(kb + ks * 32 + 20 + khalf);
            v16bf bf;
            cvt4_scaled(bf, 0,  k0v, 1.f);
            cvt4_scaled(bf, 4,  k1v, 1.f);
            cvt4_scaled(bf, 8,  k2v, 1.f);
            cvt4_scaled(bf, 12, k3v, 1.f);
            acc = __builtin_amdgcn_wmma_f32_16x16x32_bf16(
                false, qa[ks], false, bf, (short)0, acc, false, false);
        }
        const int nn = n0 + (lane & 15);
#pragma unroll
        for (int i = 0; i < 8; ++i)
            sS[i + ((lane & 16) ? 8 : 0)][nn] = acc[i];
    }
    __syncthreads();

    // ---- Phase 2: softmax along 577 cols (wave32 shfl reductions) ----
    for (int r = wv; r < 16; r += 4) {
        float mx = -1e30f;
        for (int c = lane; c < LSEQ; c += 32) mx = fmaxf(mx, sS[r][c]);
        for (int off = 16; off > 0; off >>= 1) mx = fmaxf(mx, __shfl_xor(mx, off, 32));
        float sum = 0.f;
        for (int c = lane; c < LSEQ; c += 32) {
            const float e = __expf(sS[r][c] - mx);
            sS[r][c] = e;
            sum += e;
        }
        for (int off = 16; off > 0; off >>= 1) sum += __shfl_xor(sum, off, 32);
        const float inv = 1.f / sum;
        for (int c = lane; c < LSEQ; c += 32) sS[r][c] *= inv;
        for (int c = LSEQ + lane; c < 616; c += 32) sS[r][c] = 0.f;  // zero pad
    }
    __syncthreads();

    // ---- Phase 3: O = P V ; wave wv owns d-chunk [wv*16, wv*16+16) ----
    v8f oacc = {};
    const int nd = wv * 16 + (lane & 15);
    const float* vbase = qkv + (2 * N_HEAD + h) * HDIM + nd;
    for (int kt = 0; kt < 19; ++kt) {      // ceil(577/32) steps over s
        const int s0 = kt * 32;
        // P fragment from LDS (float4 pairs, then cvt)
        v16bf pa;
        {
            const float* pr = &sS[lane & 15][s0];
            const float4 p0 = *reinterpret_cast<const float4*>(pr + khalf);
            const float4 p1 = *reinterpret_cast<const float4*>(pr + khalf + 4);
            const float4 p2 = *reinterpret_cast<const float4*>(pr + 16 + khalf);
            const float4 p3 = *reinterpret_cast<const float4*>(pr + 20 + khalf);
            cvt4_scaled(pa, 0,  p0, 1.f);
            cvt4_scaled(pa, 4,  p1, 1.f);
            cvt4_scaled(pa, 8,  p2, 1.f);
            cvt4_scaled(pa, 12, p3, 1.f);
        }
        // V fragment: column gather, unconditional clamped rows
        v16bf vb;
#pragma unroll
        for (int j = 0; j < 8; ++j) {
            const int s = min(s0 + khalf + j, LSEQ - 1);
            vb[j] = (__bf16)vbase[(size_t)(b * LSEQ + s) * 3 * D_MODEL];
        }
#pragma unroll
        for (int j = 0; j < 8; ++j) {
            const int s = min(s0 + 16 + khalf + j, LSEQ - 1);
            vb[8 + j] = (__bf16)vbase[(size_t)(b * LSEQ + s) * 3 * D_MODEL];
        }
        oacc = __builtin_amdgcn_wmma_f32_16x16x32_bf16(
            false, pa, false, vb, (short)0, oacc, false, false);
    }
#pragma unroll
    for (int i = 0; i < 8; ++i) {
        const int l = l0 + i + ((lane & 16) ? 8 : 0);
        if (l < LSEQ)
            o[(size_t)(b * LSEQ + l) * D_MODEL + h * HDIM + nd] = oacc[i];
    }
}

// ---------------------------------------------------------------------------
// LayerNorm: one block (256 thr) per 1024-wide row.
// ---------------------------------------------------------------------------
__global__ void __launch_bounds__(256)
ln_kernel(const float* __restrict__ x, const float* __restrict__ g,
          const float* __restrict__ bt, float* __restrict__ out)
{
    const int row  = blockIdx.x;
    const int base = threadIdx.x * 4;
    const float4 v = *reinterpret_cast<const float4*>(x + (size_t)row * D_MODEL + base);
    float s  = v.x + v.y + v.z + v.w;
    float s2 = v.x * v.x + v.y * v.y + v.z * v.z + v.w * v.w;
    for (int off = 16; off > 0; off >>= 1) {
        s  += __shfl_xor(s,  off, 32);
        s2 += __shfl_xor(s2, off, 32);
    }
    __shared__ float ws[8], ws2[8];
    const int wv = threadIdx.x >> 5, lane = threadIdx.x & 31;
    if (lane == 0) { ws[wv] = s; ws2[wv] = s2; }
    __syncthreads();
    if (wv == 0) {
        float a  = (lane < 8) ? ws[lane]  : 0.f;
        float b2 = (lane < 8) ? ws2[lane] : 0.f;
        for (int off = 4; off > 0; off >>= 1) {
            a  += __shfl_xor(a,  off, 32);
            b2 += __shfl_xor(b2, off, 32);
        }
        if (lane == 0) { ws[0] = a; ws2[0] = b2; }
    }
    __syncthreads();
    const float mean = ws[0] * (1.f / D_MODEL);
    const float var  = ws2[0] * (1.f / D_MODEL) - mean * mean;
    const float inv  = rsqrtf(var + 1e-5f);
    const float4 gv = *reinterpret_cast<const float4*>(g + base);
    const float4 bv = *reinterpret_cast<const float4*>(bt + base);
    float4 ov;
    ov.x = (v.x - mean) * inv * gv.x + bv.x;
    ov.y = (v.y - mean) * inv * gv.y + bv.y;
    ov.z = (v.z - mean) * inv * gv.z + bv.z;
    ov.w = (v.w - mean) * inv * gv.w + bv.w;
    *reinterpret_cast<float4*>(out + (size_t)row * D_MODEL + base) = ov;
}

// ---------------------------------------------------------------------------
// Elementwise helpers
// ---------------------------------------------------------------------------
__global__ void copy4_kernel(const float4* __restrict__ in, float4* __restrict__ out, int n4)
{
    const int i = blockIdx.x * 256 + threadIdx.x;
    if (i < n4) out[i] = in[i];
}

// h_lin [18432,128] -> conv layout [B,Ca,T,H,W]
__global__ void to_conv_kernel(const float* __restrict__ hl, float* __restrict__ hc)
{
    const int idx = blockIdx.x * 256 + threadIdx.x;
    if (idx >= NCONV) return;
    int t1 = idx;
    const int x2 = t1 % CHW; t1 /= CHW;
    const int y  = t1 % CHW; t1 /= CHW;
    const int t  = t1 % CT;  t1 /= CT;
    const int ca = t1 % CA;
    const int b  = t1 / CA;
    hc[idx] = hl[(size_t)((b * CT + t) * LM1 + y * CHW + x2) * CA + ca];
}

// conv layout -> [18432,128]
__global__ void from_conv_kernel(const float* __restrict__ hc, float* __restrict__ hl)
{
    const int idx = blockIdx.x * 256 + threadIdx.x;
    if (idx >= NCONV) return;
    int t1 = idx;
    const int ca = t1 % CA; t1 /= CA;
    const int p  = t1 % LM1;
    const int bt = t1 / LM1;
    const int b  = bt / CT, t = bt % CT;
    const int y  = p / CHW, x2 = p % CHW;
    hl[idx] = hc[(size_t)(((b * CA + ca) * CT + t) * CHW + y) * CHW + x2];
}

__device__ __forceinline__ float conv_at(const float* base, int tt, int yy, int xx)
{
    if (tt < 0 || tt >= CT || yy < 0 || yy >= CHW || xx < 0 || xx >= CHW) return 0.f;
    return base[(tt * CHW + yy) * CHW + xx];
}

// c = (conv333 + conv311 + conv133)/3 + h   (depthwise, zero pad)
__global__ void convsum_kernel(const float* __restrict__ hin,
                               const float* __restrict__ w1, const float* __restrict__ b1,
                               const float* __restrict__ w2, const float* __restrict__ b2,
                               const float* __restrict__ w3, const float* __restrict__ b3,
                               float* __restrict__ out)
{
    const int idx = blockIdx.x * 256 + threadIdx.x;
    if (idx >= NCONV) return;
    int t1 = idx;
    const int x2 = t1 % CHW; t1 /= CHW;
    const int y  = t1 % CHW; t1 /= CHW;
    const int t  = t1 % CT;  t1 /= CT;
    const int ca = t1 % CA;
    const int b  = t1 / CA;
    const float* base = hin + (size_t)((b * CA + ca) * CT) * CHW * CHW;
    float c1 = 0.f;
#pragma unroll
    for (int dt = -1; dt <= 1; ++dt)
#pragma unroll
        for (int dy = -1; dy <= 1; ++dy)
#pragma unroll
            for (int dx = -1; dx <= 1; ++dx)
                c1 += conv_at(base, t + dt, y + dy, x2 + dx) *
                      w1[ca * 27 + (dt + 1) * 9 + (dy + 1) * 3 + (dx + 1)];
    float c2 = 0.f;
#pragma unroll
    for (int dt = -1; dt <= 1; ++dt)
        c2 += conv_at(base, t + dt, y, x2) * w2[ca * 3 + dt + 1];
    float c3 = 0.f;
#pragma unroll
    for (int dy = -1; dy <= 1; ++dy)
#pragma unroll
        for (int dx = -1; dx <= 1; ++dx)
            c3 += conv_at(base, t, y + dy, x2 + dx) * w3[ca * 9 + (dy + 1) * 3 + (dx + 1)];
    const float ctr = conv_at(base, t, y, x2);
    out[idx] = (c1 + b1[ca] + c2 + b2[ca] + c3 + b3[ca]) * (1.f / 3.f) + ctr;
}

// out = in + conv333(in, pw) + pb
__global__ void convproj_kernel(const float* __restrict__ cin,
                                const float* __restrict__ pw, const float* __restrict__ pb,
                                float* __restrict__ out)
{
    const int idx = blockIdx.x * 256 + threadIdx.x;
    if (idx >= NCONV) return;
    int t1 = idx;
    const int x2 = t1 % CHW; t1 /= CHW;
    const int y  = t1 % CHW; t1 /= CHW;
    const int t  = t1 % CT;  t1 /= CT;
    const int ca = t1 % CA;
    const int b  = t1 / CA;
    const float* base = cin + (size_t)((b * CA + ca) * CT) * CHW * CHW;
    float c = 0.f;
#pragma unroll
    for (int dt = -1; dt <= 1; ++dt)
#pragma unroll
        for (int dy = -1; dy <= 1; ++dy)
#pragma unroll
            for (int dx = -1; dx <= 1; ++dx)
                c += conv_at(base, t + dt, y + dy, x2 + dx) *
                     pw[ca * 27 + (dt + 1) * 9 + (dy + 1) * 3 + (dx + 1)];
    out[idx] = conv_at(base, t, y, x2) + c + pb[ca];
}

// ---------------------------------------------------------------------------
// Host launcher
// ---------------------------------------------------------------------------
extern "C" void kernel_launch(void* const* d_in, const int* in_sizes, int n_in,
                              void* d_out, int out_size, void* d_ws, size_t ws_size,
                              hipStream_t stream)
{
    const float* x_in  = (const float*)d_in[0];
    const float* in_w  = (const float*)d_in[1];
    const float* in_b  = (const float*)d_in[2];
    const float* out_w = (const float*)d_in[3];
    const float* out_b = (const float*)d_in[4];
    const float* fc_w  = (const float*)d_in[5];
    const float* fc_b  = (const float*)d_in[6];
    const float* pj_w  = (const float*)d_in[7];
    const float* pj_b  = (const float*)d_in[8];
    const float* ln1_g = (const float*)d_in[9];
    const float* ln1_b = (const float*)d_in[10];
    const float* ln2_g = (const float*)d_in[11];
    const float* ln2_b = (const float*)d_in[12];
    const float* const* ad1 = (const float* const*)&d_in[13];  // 12 entries
    const float* const* ad2 = (const float* const*)&d_in[25];

    constexpr size_t NX   = (size_t)BT * LSEQ * D_MODEL;      // 18,907,136
    constexpr size_t NBIG = (size_t)BT * LSEQ * 4 * D_MODEL;  // 75,628,544
    constexpr size_t NHID = (size_t)M_AD * CA;                // 2,359,296

    float* ws     = (float*)d_ws;
    float* buf_x  = ws;                 // running x
    float* ln_out = ws + NX;            // LN output / attention output (aliased)
    float* att    = ln_out;
    float* big    = ws + 2 * NX;        // qkv / mlp hidden
    float* h_lin  = big + NBIG;         // adapter hidden [18432,128]
    float* cb1    = h_lin + NHID;       // conv scratch
    float* cb2    = cb1 + NHID;

    const int cgrid = (NCONV + 255) / 256;

    // buf_x = x  (vectorized)
    copy4_kernel<<<(int)((NX / 4 + 255) / 256), 256, 0, stream>>>(
        (const float4*)x_in, (float4*)buf_x, (int)(NX / 4));

    auto run_adapter = [&](const float* const* p) {
        const float *fc1w = p[0], *fc1b = p[1], *fc2w = p[2], *fc2b = p[3];
        const float *c1w = p[4], *c1b = p[5], *c2w = p[6], *c2b = p[7];
        const float *c3w = p[8], *c3b = p[9], *prw = p[10], *prb = p[11];
        // h = x[:,1:,:] @ fc1^T + b   (AMAP=1 skips CLS row)
        gemm_wmma_kernel<1, 0, 0, 0><<<dim3(CA / 64, M_AD / 128), 256, 0, stream>>>(
            buf_x, fc1w, fc1b, nullptr, h_lin, M_AD, CA, D_MODEL);
        to_conv_kernel<<<cgrid, 256, 0, stream>>>(h_lin, cb1);
        convsum_kernel<<<cgrid, 256, 0, stream>>>(cb1, c1w, c1b, c2w, c2b, c3w, c3b, cb2);
        convproj_kernel<<<cgrid, 256, 0, stream>>>(cb2, prw, prb, cb1);
        from_conv_kernel<<<cgrid, 256, 0, stream>>>(cb1, h_lin);
        // x[:,1:,:] += c @ fc2^T + b   (SMAP=1 scatter, RES adds current x)
        gemm_wmma_kernel<0, 1, 0, 1><<<dim3(D_MODEL / 64, M_AD / 128), 256, 0, stream>>>(
            h_lin, fc2w, fc2b, buf_x, buf_x, M_AD, D_MODEL, CA);
    };

    const int mtiles = (M_FULL + 127) / 128;   // 145

    // ---- adapter 1 ----
    run_adapter(ad1);

    // ---- x = x + attn(LN1(x)) ----
    ln_kernel<<<M_FULL, 256, 0, stream>>>(buf_x, ln1_g, ln1_b, ln_out);
    gemm_wmma_kernel<0, 0, 0, 0><<<dim3(3 * D_MODEL / 64, mtiles), 256, 0, stream>>>(
        ln_out, in_w, in_b, nullptr, big, M_FULL, 3 * D_MODEL, D_MODEL);
    attn_kernel<<<dim3((LSEQ + 15) / 16, N_HEAD, BT), 128, 0, stream>>>(big, att);
    gemm_wmma_kernel<0, 0, 0, 1><<<dim3(D_MODEL / 64, mtiles), 256, 0, stream>>>(
        att, out_w, out_b, buf_x, buf_x, M_FULL, D_MODEL, D_MODEL);

    // ---- adapter 2 ----
    run_adapter(ad2);

    // ---- MLP: out = x + proj(QuickGELU(fc(LN2(x)))) ----
    ln_kernel<<<M_FULL, 256, 0, stream>>>(buf_x, ln2_g, ln2_b, ln_out);
    gemm_wmma_kernel<0, 0, 1, 0><<<dim3(4 * D_MODEL / 64, mtiles), 256, 0, stream>>>(
        ln_out, fc_w, fc_b, nullptr, big, M_FULL, 4 * D_MODEL, D_MODEL);
    gemm_wmma_kernel<0, 0, 0, 1><<<dim3(D_MODEL / 64, mtiles), 256, 0, stream>>>(
        big, pj_w, pj_b, buf_x, (float*)d_out, M_FULL, D_MODEL, 4 * D_MODEL);
}
